// ScoreLoss_69715909149246
// MI455X (gfx1250) — compile-verified, matrix operand order checked by reading, not run on hardware
//
#include <hip/hip_runtime.h>
#include <hip/hip_bf16.h>

// Rank-count via CDNA5 matrix pipe:
//   D = A(16x64 iu8 {0,1} comparison tile) x B(64x16 iu8 all-ones) + C(i32)
// Every column of D holds the row sum => sa/sb accumulate in 8 VGPRs, no
// cross-lane reduction needed until the very end. v8i = 8 x i32 (wave32).
typedef __attribute__((ext_vector_type(8))) int v8i;

#define JT 256          // j super-tile staged in LDS (4 WMMA K-blocks of 64)
#define WAVES_PER_BLOCK 4

// 128 threads = 4 waves per block; each wave owns one 16-row output tile.
// Block cooperatively stages JT j-values of pred+gt per iteration; all 4
// waves consume the same LDS tile (broadcast reads).
__global__ __launch_bounds__(128) void ScoreLoss_rank_wmma_kernel(
    const float* __restrict__ pred, const float* __restrict__ gt,
    float* __restrict__ ws, int N) {
  __shared__ __align__(16) float lp[JT];
  __shared__ __align__(16) float lg[JT];

  const int t    = threadIdx.x;
  const int lane = t & 31;               // lane within wave32
  const int wave = t >> 5;               // 0..3
  const int hi   = lane >> 4;            // half-wave select
  const int m    = lane & 15;            // row within 16-row tile
  const int tile = blockIdx.x * WAVES_PER_BLOCK + wave;
  const int row  = tile * 16 + m;

  const float pv = pred[row];
  const float gv = gt[row];

  // All-ones B matrix (64x16 iu8): every byte == 1, layout-independent.
  v8i ones;
#pragma unroll
  for (int r = 0; r < 8; ++r) ones[r] = 0x01010101;

  v8i accP = {};  // sa accumulator (16x16 i32 C/D)
  v8i accG = {};  // sb accumulator

  // K base per A-operand VGPR (8-bit 16x64 A layout):
  // VGPR0: K0-3 / K8-11(hi), VGPR1: K4-7 / K12-15, VGPR2: +16, VGPR4: +32 ...
  int kb[8];
#pragma unroll
  for (int v = 0; v < 8; ++v)
    kb[v] = ((v & 1) * 4) + (((v >> 1) & 1) * 16) + (((v >> 2) & 1) * 32) + hi * 8;

  for (int j0 = 0; j0 < N; j0 += JT) {
    // Stage JT j-values of each array (128 threads x 2 elements each).
    lp[t]       = pred[j0 + t];
    lp[t + 128] = pred[j0 + 128 + t];
    lg[t]       = gt[j0 + t];
    lg[t + 128] = gt[j0 + 128 + t];
    __syncthreads();

#pragma unroll
    for (int kk = 0; kk < JT / 64; ++kk) {
      const int base = kk * 64;
      v8i aP, aG;
#pragma unroll
      for (int v = 0; v < 8; ++v) {
        const float4 fp = *(const float4*)&lp[base + kb[v]];  // ds_load_b128
        const float4 fg = *(const float4*)&lg[base + kb[v]];
        unsigned bp = (unsigned)(pv > fp.x)         | ((unsigned)(pv > fp.y) << 8) |
                      ((unsigned)(pv > fp.z) << 16) | ((unsigned)(pv > fp.w) << 24);
        unsigned bg = (unsigned)(gv > fg.x)         | ((unsigned)(gv > fg.y) << 8) |
                      ((unsigned)(gv > fg.z) << 16) | ((unsigned)(gv > fg.w) << 24);
        aP[v] = (int)bp;
        aG[v] = (int)bg;
      }
      // 7-arg iu8 WMMA: (sgn_a, A, sgn_b, B, C, reuse_a, reuse_b).
      accP = __builtin_amdgcn_wmma_i32_16x16x64_iu8(false, aP, false, ones,
                                                    accP, false, false);
      accG = __builtin_amdgcn_wmma_i32_16x16x64_iu8(false, aG, false, ones,
                                                    accG, false, false);
    }
    __syncthreads();  // protect LDS super-tile before next stage
  }

  // Lanes 0-15 hold rows 0-7 (VGPR r -> row r), lanes 16-31 rows 8-15; columns
  // are duplicates since B is all-ones. Sum |sa - sb| for this wave's rows.
  float s = 0.f;
#pragma unroll
  for (int r = 0; r < 8; ++r) s += fabsf((float)(accP[r] - accG[r]));

  float other = __shfl_xor(s, 16, 32);   // combine the two half-wave row groups
  if (lane == 0) ws[tile] = s + other;
}

// Deterministic final reduction: sum per-tile partials, scale, write scalar.
__global__ void ScoreLoss_reduce_kernel(const float* __restrict__ ws,
                                        float* __restrict__ out, int ntiles,
                                        float scale) {
  __shared__ float sm[256];
  const int t = threadIdx.x;
  float s = 0.f;
  for (int i = t; i < ntiles; i += 256) s += ws[i];
  sm[t] = s;
  __syncthreads();
  for (int off = 128; off > 0; off >>= 1) {
    if (t < off) sm[t] += sm[t + off];
    __syncthreads();
  }
  if (t == 0) out[0] = sm[0] * scale;
}

extern "C" void kernel_launch(void* const* d_in, const int* in_sizes, int n_in,
                              void* d_out, int out_size, void* d_ws, size_t ws_size,
                              hipStream_t stream) {
  const float* pred = (const float*)d_in[0];
  const float* gt   = (const float*)d_in[1];
  const int N = in_sizes[0];                  // 8192; divisible by JT
  float* ws = (float*)d_ws;                   // tiles * 4 B = 2 KB scratch
  const int tiles  = N / 16;                  // 512 waves
  const int blocks = tiles / WAVES_PER_BLOCK; // 128 blocks x 128 threads

  ScoreLoss_rank_wmma_kernel<<<blocks, 128, 0, stream>>>(pred, gt, ws, N);

  // loss = sum(|sa - sb|) / ((N + 1e-8) * N)
  const double scale = 1.0 / (((double)N + 1e-8) * (double)N);
  ScoreLoss_reduce_kernel<<<1, 256, 0, stream>>>(ws, (float*)d_out, tiles,
                                                 (float)scale);
}
